// Decoder_88493506167346
// MI455X (gfx1250) — matrix operand order repaired and checked by reading, not compiled, not run on hardware
//
#include <hip/hip_runtime.h>
#include <math.h>

// ---------------------------------------------------------------------------
// GRU decoder, teacher forcing.  T=50, B=64, E=H=256, V=32000.
// Heavy op: per-step (64 x 32000) projection -> bf16 WMMA with W_out tile
// staged into LDS via gfx1250 async global->LDS loads (ASYNCcnt), shared by
// all 4 waves of the block.  fp32 accumulation, fp32 recurrent state.
// ---------------------------------------------------------------------------

typedef __attribute__((ext_vector_type(16))) __bf16 v16bf;
typedef __attribute__((ext_vector_type(8)))  __bf16 v8bf;
typedef __attribute__((ext_vector_type(8)))  float  v8f;
typedef int v4i __attribute__((vector_size(16)));          // matches builtin param type
typedef __attribute__((address_space(1))) v4i* gptr_v4i;   // global (AS1)
typedef __attribute__((address_space(3))) v4i* lptr_v4i;   // LDS (AS3)

#define SOS_TOKEN 0
constexpr int Vv = 32000;
constexpr int Ee = 256;
constexpr int Hh = 256;
constexpr int Bb = 64;
constexpr int Tt = 50;
constexpr int G3 = 3 * Hh;           // 768
constexpr int LDSW = 264;            // padded LDS row stride in halfs (528B = 132 dw -> +4 banks/row)

#if __has_builtin(__builtin_amdgcn_global_load_async_to_lds_b128)
#define HAVE_ASYNC_LDS 1
#else
#define HAVE_ASYNC_LDS 0
#endif

__device__ inline void wait_async_zero() {
#if __has_builtin(__builtin_amdgcn_s_wait_asynccnt)
    __builtin_amdgcn_s_wait_asynccnt(0);
#else
    asm volatile("s_wait_asynccnt 0" ::: "memory");
#endif
}

// ---- WMMA fragment helpers (layouts per CDNA5 ISA 7.12.2, wave32) ----------

// A (16x32 bf16, row-major src, leading dim lda in elements)
__device__ inline v16bf load_a_frag(const __bf16* A, int lda, int mBase, int kk, int lane) {
    int row = mBase + (lane & 15);
    int kb  = kk + ((lane >> 4) << 3);            // +0 or +8
    const __bf16* p = A + (size_t)row * lda + kb;
    v8bf lo = *(const v8bf*)(p);                  // K = kb .. kb+7
    v8bf hi = *(const v8bf*)(p + 16);             // K = kb+16 .. kb+23
    v16bf a;
#pragma unroll
    for (int i = 0; i < 8; ++i) { a[i] = lo[i]; a[i + 8] = hi[i]; }
    return a;
}

// B (32x16 bf16) as transpose of row-major W (N x K, leading dim ldw):
// lane holds column n = nBase + (lane&15), K = kb..kb+7 and kb+16..kb+23, kb = kk + (lane>>4)*8
__device__ inline v16bf load_b_frag(const __bf16* W, int ldw, int nBase, int kk, int lane) {
    int col = nBase + (lane & 15);
    int kb  = kk + ((lane >> 4) << 3);            // +0 or +8
    const __bf16* p = W + (size_t)col * ldw + kb;
    v8bf lo = *(const v8bf*)(p);
    v8bf hi = *(const v8bf*)(p + 16);
    v16bf b;
#pragma unroll
    for (int i = 0; i < 8; ++i) { b[i] = lo[i]; b[i + 8] = hi[i]; }
    return b;
}

// C/D (16x16 f32): lane<16 -> N=lane, M=vgpr; lane>=16 -> N=lane-16, M=vgpr+8
__device__ inline void store_tile(float* C, int ldc, int mBase, int nBase,
                                  const v8f acc, int lane, const float* bias) {
    int n  = nBase + (lane & 15);
    int m0 = mBase + ((lane >> 4) << 3);
    float bb = bias ? bias[n] : 0.0f;
#pragma unroll
    for (int i = 0; i < 8; ++i)
        C[(size_t)(m0 + i) * ldc + n] = acc[i] + bb;
}

// ---- small utility kernels -------------------------------------------------

__global__ void f32_to_bf16_kernel(const float* __restrict__ in, __bf16* __restrict__ out, int n) {
    int i = blockIdx.x * blockDim.x + threadIdx.x;
    if (i < n) out[i] = (__bf16)in[i];
}

__global__ void init_h_kernel(const float* __restrict__ hidden, float* __restrict__ h,
                              __bf16* __restrict__ hb) {
    int i = blockIdx.x * blockDim.x + threadIdx.x;   // B*H = 16384
    float v = hidden[i];
    h[i]  = v;
    hb[i] = (__bf16)v;
}

// x = relu(embedding[token]) -> bf16.  grid = B blocks, block = E threads
__global__ void embed_kernel(const int* __restrict__ toks, const float* __restrict__ emb,
                             __bf16* __restrict__ xb, int di) {
    int b = blockIdx.x;
    int e = threadIdx.x;
    int tok = (di == 0) ? SOS_TOKEN : toks[(size_t)(di - 1) * Bb + b];
    float v = emb[(size_t)tok * Ee + e];
    xb[(size_t)b * Ee + e] = (__bf16)(v > 0.0f ? v : 0.0f);
}

// ---- GRU gate GEMMs: gi = x @ W_ih^T, gh = h @ W_hh^T ----------------------
// grid = 48 blocks (N tiles), block = 128 (4 waves = 4 M tiles)
__global__ __launch_bounds__(128) void gate_kernel(const __bf16* __restrict__ xb,
                                                   const __bf16* __restrict__ hb,
                                                   const __bf16* __restrict__ Wih,
                                                   const __bf16* __restrict__ Whh,
                                                   float* __restrict__ gi,
                                                   float* __restrict__ gh) {
    int lane  = threadIdx.x & 31;
    int mTile = threadIdx.x >> 5;          // 0..3 -> M = 0..63
    int nBase = blockIdx.x * 16;           // 0..767
    v8f accI = {};
    v8f accH = {};
#pragma unroll
    for (int kk = 0; kk < Ee; kk += 32) {
        v16bf ax = load_a_frag(xb, Ee, mTile * 16, kk, lane);
        v16bf ah = load_a_frag(hb, Hh, mTile * 16, kk, lane);
        v16bf bi = load_b_frag(Wih, Ee, nBase, kk, lane);
        v16bf bh = load_b_frag(Whh, Hh, nBase, kk, lane);
        accI = __builtin_amdgcn_wmma_f32_16x16x32_bf16(false, ax, false, bi, (short)0, accI, false, false);
        accH = __builtin_amdgcn_wmma_f32_16x16x32_bf16(false, ah, false, bh, (short)0, accH, false, false);
    }
    store_tile(gi, G3, mTile * 16, nBase, accI, lane, nullptr);
    store_tile(gh, G3, mTile * 16, nBase, accH, lane, nullptr);
}

// ---- GRU elementwise update ------------------------------------------------
__global__ void gru_elem_kernel(const float* __restrict__ gi, const float* __restrict__ gh,
                                const float* __restrict__ b_ih, const float* __restrict__ b_hh,
                                float* __restrict__ h, __bf16* __restrict__ hb) {
    int b = blockIdx.x;
    int j = threadIdx.x;
    const float* gib = gi + (size_t)b * G3;
    const float* ghb = gh + (size_t)b * G3;
    float ir = gib[j]          + b_ih[j];
    float iz = gib[Hh + j]     + b_ih[Hh + j];
    float in = gib[2 * Hh + j] + b_ih[2 * Hh + j];
    float hr = ghb[j]          + b_hh[j];
    float hz = ghb[Hh + j]     + b_hh[Hh + j];
    float hn = ghb[2 * Hh + j] + b_hh[2 * Hh + j];
    float r = 1.0f / (1.0f + __expf(-(ir + hr)));
    float z = 1.0f / (1.0f + __expf(-(iz + hz)));
    float n = tanhf(in + r * hn);
    size_t idx = (size_t)b * Hh + j;
    float ho   = h[idx];
    float hnew = (1.0f - z) * n + z * ho;
    h[idx]  = hnew;
    hb[idx] = (__bf16)hnew;
}

// ---- Output projection: logits = h @ W_out^T + b_out -----------------------
// grid = V/64 = 500 blocks, block = 128 (4 waves = 4 M tiles), 16x64 per wave.
// The block's 64x256 bf16 W_out tile is staged once into LDS (async DMA),
// then shared by all 4 waves; padded stride makes B-frag reads bank-free.
__global__ __launch_bounds__(128) void proj_kernel(const __bf16* __restrict__ hb,
                                                   const __bf16* __restrict__ Wout,
                                                   const float* __restrict__ bout,
                                                   float* __restrict__ logits) {
    __shared__ __bf16 wt[64 * LDSW];               // 33,792 B
    int tid   = threadIdx.x;
    int nBase = blockIdx.x * 64;
    const __bf16* gsrc = Wout + (size_t)nBase * Hh;

    // cooperative tile copy: 64 rows x 512B, 16B chunks, 16 chunks/thread
#pragma unroll
    for (int i = 0; i < 16; ++i) {
        int c   = tid + i * 128;
        int row = c >> 5;                          // 0..63
        int off = (c & 31) * 8;                    // half offset, 16B chunks
#if HAVE_ASYNC_LDS
        __builtin_amdgcn_global_load_async_to_lds_b128(
            (gptr_v4i)(gsrc + (size_t)row * Hh + off),
            (lptr_v4i)(wt + row * LDSW + off),
            0, 0);
#else
        *(v8bf*)(wt + row * LDSW + off) = *(const v8bf*)(gsrc + (size_t)row * Hh + off);
#endif
    }
#if HAVE_ASYNC_LDS
    wait_async_zero();
#endif
    __syncthreads();

    int lane  = tid & 31;
    int mTile = tid >> 5;                          // 0..3
    v8f acc0 = {}, acc1 = {}, acc2 = {}, acc3 = {};
#pragma unroll
    for (int kk = 0; kk < Hh; kk += 32) {
        v16bf a  = load_a_frag(hb, Hh, mTile * 16, kk, lane);
        v16bf b0 = load_b_frag(wt, LDSW, 0,  kk, lane);
        v16bf b1 = load_b_frag(wt, LDSW, 16, kk, lane);
        v16bf b2 = load_b_frag(wt, LDSW, 32, kk, lane);
        v16bf b3 = load_b_frag(wt, LDSW, 48, kk, lane);
        acc0 = __builtin_amdgcn_wmma_f32_16x16x32_bf16(false, a, false, b0, (short)0, acc0, false, false);
        acc1 = __builtin_amdgcn_wmma_f32_16x16x32_bf16(false, a, false, b1, (short)0, acc1, false, false);
        acc2 = __builtin_amdgcn_wmma_f32_16x16x32_bf16(false, a, false, b2, (short)0, acc2, false, false);
        acc3 = __builtin_amdgcn_wmma_f32_16x16x32_bf16(false, a, false, b3, (short)0, acc3, false, false);
    }
    store_tile(logits, Vv, mTile * 16, nBase,      acc0, lane, bout);
    store_tile(logits, Vv, mTile * 16, nBase + 16, acc1, lane, bout);
    store_tile(logits, Vv, mTile * 16, nBase + 32, acc2, lane, bout);
    store_tile(logits, Vv, mTile * 16, nBase + 48, acc3, lane, bout);
}

// ---- log-softmax over V per row, write to output ---------------------------
__global__ __launch_bounds__(256) void logsoftmax_kernel(const float* __restrict__ logits,
                                                         float* __restrict__ out, int di) {
    __shared__ float red[256];
    int b = blockIdx.x;
    int t = threadIdx.x;
    const float* row = logits + (size_t)b * Vv;

    float m = -1e30f;
    for (int v = t; v < Vv; v += 256) m = fmaxf(m, row[v]);
    red[t] = m; __syncthreads();
    for (int s = 128; s > 0; s >>= 1) { if (t < s) red[t] = fmaxf(red[t], red[t + s]); __syncthreads(); }
    float rmax = red[0]; __syncthreads();

    float sum = 0.0f;
    for (int v = t; v < Vv; v += 256) sum += __expf(row[v] - rmax);
    red[t] = sum; __syncthreads();
    for (int s = 128; s > 0; s >>= 1) { if (t < s) red[t] += red[t + s]; __syncthreads(); }
    float lse = rmax + logf(red[0]);

    float* orow = out + ((size_t)di * Bb + b) * Vv;
    for (int v = t; v < Vv; v += 256) orow[v] = row[v] - lse;
}

// ---------------------------------------------------------------------------

extern "C" void kernel_launch(void* const* d_in, const int* in_sizes, int n_in,
                              void* d_out, int out_size, void* d_ws, size_t ws_size,
                              hipStream_t stream) {
    const float* hidden = (const float*)d_in[0];   // (1,B,H)
    const int*   toks   = (const int*)  d_in[1];   // (T,B)
    const float* emb    = (const float*)d_in[2];   // (V,E)
    const float* W_ih   = (const float*)d_in[3];   // (3H,E)
    const float* W_hh   = (const float*)d_in[4];   // (3H,H)
    const float* b_ih   = (const float*)d_in[5];
    const float* b_hh   = (const float*)d_in[6];
    const float* W_out  = (const float*)d_in[7];   // (V,H)
    const float* b_out  = (const float*)d_in[8];
    float* out = (float*)d_out;                    // T*B*V logprobs ++ B*H h_final

    char*  ws  = (char*)d_ws;
    size_t off = 0;
    auto carve = [&](size_t bytes) -> void* {
        void* p = ws + off;
        off = (off + bytes + 255) & ~(size_t)255;
        return p;
    };
    __bf16* Wout_b = (__bf16*)carve((size_t)Vv * Hh * 2);   // 16.38 MB
    __bf16* Wih_b  = (__bf16*)carve((size_t)G3 * Ee * 2);
    __bf16* Whh_b  = (__bf16*)carve((size_t)G3 * Hh * 2);
    __bf16* xb     = (__bf16*)carve((size_t)Bb * Ee * 2);
    __bf16* hb     = (__bf16*)carve((size_t)Bb * Hh * 2);
    float*  hbuf   = (float*) carve((size_t)Bb * Hh * 4);
    float*  gi     = (float*) carve((size_t)Bb * G3 * 4);
    float*  gh     = (float*) carve((size_t)Bb * G3 * 4);
    float*  logits = (float*) carve((size_t)Bb * Vv * 4);   // 8.19 MB

    f32_to_bf16_kernel<<<(Vv * Hh + 255) / 256, 256, 0, stream>>>(W_out, Wout_b, Vv * Hh);
    f32_to_bf16_kernel<<<(G3 * Ee + 255) / 256, 256, 0, stream>>>(W_ih, Wih_b, G3 * Ee);
    f32_to_bf16_kernel<<<(G3 * Hh + 255) / 256, 256, 0, stream>>>(W_hh, Whh_b, G3 * Hh);
    init_h_kernel<<<(Bb * Hh) / 256, 256, 0, stream>>>(hidden, hbuf, hb);

    for (int di = 0; di < Tt; ++di) {
        embed_kernel<<<Bb, Ee, 0, stream>>>(toks, emb, xb, di);
        gate_kernel<<<G3 / 16, 128, 0, stream>>>(xb, hb, Wih_b, Whh_b, gi, gh);
        gru_elem_kernel<<<Bb, Hh, 0, stream>>>(gi, gh, b_ih, b_hh, hbuf, hb);
        proj_kernel<<<Vv / 64, 128, 0, stream>>>(hb, Wout_b, b_out, logits);
        logsoftmax_kernel<<<Bb, 256, 0, stream>>>(logits, out, di);
    }

    (void)hipMemcpyAsync(out + (size_t)Tt * Bb * Vv, hbuf, (size_t)Bb * Hh * sizeof(float),
                         hipMemcpyDeviceToDevice, stream);
}
